// TimeSeriesSelfAttention_83253646065736
// MI455X (gfx1250) — matrix operand order repaired and checked by reading, not compiled, not run on hardware
//
#include <hip/hip_runtime.h>
#include <hip/hip_bf16.h>
#include <math.h>

typedef _Float16 f16;
typedef __attribute__((ext_vector_type(16))) _Float16 v16h;
typedef __attribute__((ext_vector_type(8)))  _Float16 v8h;
typedef __attribute__((ext_vector_type(8)))  float    v8f;

#define BB 8
#define SS 2048
#define DD 512
#define RR (BB*SS)          // 16384 rows total
#define QSCALE 0.044194173824159216f   // 1/sqrt(512)

// ---------------- WMMA helpers ----------------

__device__ __forceinline__ v8f wmma16(v16h a, v16h b, v8f c) {
    // D = A(16x32 f16) * B(32x16 f16) + C(16x16 f32)
    return __builtin_amdgcn_wmma_f32_16x16x32_f16(
        false, a, false, b, (short)0, c, false, false);
}

// A-fragment (16x32, f16): lane l holds row (l&15); k-elements:
// j=0..15 -> k = half*16 + j  (half = l>>4)
// => two contiguous 8-half (16B) chunks at kb+half*16 and kb+half*16+8.
__device__ __forceinline__ v16h load_a_frag(const f16* base, size_t stride,
                                            int row, int kb, int half) {
    const f16* p = base + (size_t)row * stride + (size_t)(kb + half * 16);
    union { v16h v; v8h h[2]; } u;
    u.h[0] = *(const v8h*)p;
    u.h[1] = *(const v8h*)(p + 8);
    return u.v;
}

// B-fragment (32x16, f16): lane l holds col (l&15); k = (l>>4)*16 + j
// => one contiguous 16-half (32B) chunk. base must be row-major [col][k].
__device__ __forceinline__ v16h load_b_frag(const f16* base, size_t stride,
                                            int col, int kb, int half) {
    const f16* p = base + (size_t)col * stride + (size_t)(kb + half * 16);
    return *(const v16h*)p;
}

// ---------------- Kernel 0a: x (f32) -> xh (f16) ----------------

__global__ void cvt_x_kernel(const float* __restrict__ x, f16* __restrict__ xh, int n) {
    int i = blockIdx.x * blockDim.x + threadIdx.x;
    if (i < n) xh[i] = (f16)x[i];
}

// ---------------- Kernel 0b: W (f32,[k][n]) -> WT (f16,[n][k]) for 4 weights ----

__global__ void cvt_w_kernel(const float* __restrict__ Wq, const float* __restrict__ Wk,
                             const float* __restrict__ Wv, const float* __restrict__ Wo,
                             f16* __restrict__ WqT, f16* __restrict__ WkT,
                             f16* __restrict__ WvT, f16* __restrict__ WoT) {
    int z = blockIdx.z;
    const float* W = (z == 0) ? Wq : (z == 1) ? Wk : (z == 2) ? Wv : Wo;
    f16* WT       = (z == 0) ? WqT : (z == 1) ? WkT : (z == 2) ? WvT : WoT;
    __shared__ float tile[16][17];
    int tx = threadIdx.x, ty = threadIdx.y;
    int k0 = blockIdx.x * 16, n0 = blockIdx.y * 16;
    tile[ty][tx] = W[(size_t)(k0 + ty) * DD + (n0 + tx)];
    __syncthreads();
    WT[(size_t)(n0 + ty) * DD + (k0 + tx)] = (f16)tile[tx][ty];
}

// ---------------- Kernel 1: QKV projection GEMM -----------------
// grid (RR/64, DD/128, 3); block 256 (8 waves).
// wave w: rows mbase..mbase+15, cols nbase..nbase+63 (4 accumulators).

__global__ __launch_bounds__(256) void qkv_kernel(
        const f16* __restrict__ xh,
        const f16* __restrict__ WqT, const f16* __restrict__ WkT, const f16* __restrict__ WvT,
        const float* __restrict__ bq, const float* __restrict__ bk, const float* __restrict__ bv,
        f16* __restrict__ Qo, f16* __restrict__ Ko, f16* __restrict__ VTo) {
    const int z = blockIdx.z;
    const f16*   WT   = (z == 0) ? WqT : (z == 1) ? WkT : WvT;
    const float* bias = (z == 0) ? bq  : (z == 1) ? bk  : bv;

    int tid = threadIdx.x;
    int w = tid >> 5, l = tid & 31, half = l >> 4, ln = l & 15;
    int mbase = blockIdx.x * 64 + (w & 3) * 16;
    int nbase = blockIdx.y * 128 + (w >> 2) * 64;

    const v8f zero8 = {0.f,0.f,0.f,0.f,0.f,0.f,0.f,0.f};
    v8f acc[4] = {zero8, zero8, zero8, zero8};

    for (int kb = 0; kb < DD; kb += 32) {
        v16h A = load_a_frag(xh, DD, mbase + ln, kb, half);
        #pragma unroll
        for (int nb = 0; nb < 4; ++nb) {
            v16h Bv = load_b_frag(WT, DD, nbase + nb * 16 + ln, kb, half);
            acc[nb] = wmma16(A, Bv, acc[nb]);
        }
    }

    const float scale = (z == 0) ? QSCALE : 1.0f;
    #pragma unroll
    for (int nb = 0; nb < 4; ++nb) {
        int col = nbase + nb * 16 + ln;
        float bsv = bias[col];
        #pragma unroll
        for (int r = 0; r < 8; ++r) {
            int row = mbase + r + 8 * half;      // global row in [0,RR)
            float v = (acc[nb][r] + bsv) * scale;
            if (z == 0) {
                Qo[(size_t)row * DD + col] = (f16)v;
            } else if (z == 1) {
                Ko[(size_t)row * DD + col] = (f16)v;
            } else {
                int b = row >> 11, s = row & (SS - 1);
                VTo[((size_t)(b * DD + col)) * SS + s] = (f16)v;   // V transposed [b][d][s]
            }
        }
    }
}

// ---------------- Kernel 2: flash attention -----------------
// grid (SS/64, BB); block 256 (8 waves).
// wave w: 16 query rows (qsub = w&3), 256 output cols (dsplit = w>>2).

__global__ __launch_bounds__(256) void attn_kernel(
        const f16* __restrict__ Q, const f16* __restrict__ K,
        const f16* __restrict__ VT, f16* __restrict__ ctx) {
    __shared__ __align__(32) _Float16 lp[8][16][64];   // per-wave P scratch (16 KB)
    __shared__ float sm[8][16];
    __shared__ float sl[8][16];

    int b = blockIdx.y;
    int q0 = blockIdx.x * 64;
    int tid = threadIdx.x;
    int w = tid >> 5, l = tid & 31, half = l >> 4, ln = l & 15;
    int mrow = q0 + (w & 3) * 16;     // wave's query-row base within batch
    int d0 = (w >> 2) * 256;          // wave's output-column base

    const f16* Qb  = Q  + (size_t)b * SS * DD;
    const f16* Kb  = K  + (size_t)b * SS * DD;
    const f16* VTb = VT + (size_t)b * DD * SS;

    if (l < 16) { sm[w][l] = -3.0e38f; sl[w][l] = 0.0f; }

    const v8f zero8 = {0.f,0.f,0.f,0.f,0.f,0.f,0.f,0.f};
    v8f O[16];
    #pragma unroll
    for (int nt = 0; nt < 16; ++nt) O[nt] = zero8;

    for (int kb = 0; kb < SS; kb += 64) {
        if (kb + 64 < SS)
            __builtin_prefetch((const void*)(Kb + (size_t)(kb + 64 + ln) * DD), 0, 0);

        // ---- S = Q * K^T for 16 q-rows x 64 keys (Q already pre-scaled) ----
        v8f S[4] = {zero8, zero8, zero8, zero8};
        for (int dc = 0; dc < DD; dc += 32) {
            v16h A = load_a_frag(Qb, DD, mrow + ln, dc, half);
            #pragma unroll
            for (int nb = 0; nb < 4; ++nb) {
                v16h Bv = load_b_frag(Kb, DD, kb + nb * 16 + ln, dc, half);
                S[nb] = wmma16(A, Bv, S[nb]);
            }
        }

        // ---- online softmax update ----
        float alpha[8];
        #pragma unroll
        for (int r = 0; r < 8; ++r) {
            int row = r + 8 * half;
            float mx = fmaxf(fmaxf(S[0][r], S[1][r]), fmaxf(S[2][r], S[3][r]));
            mx = fmaxf(mx, __shfl_xor(mx, 1, 32));
            mx = fmaxf(mx, __shfl_xor(mx, 2, 32));
            mx = fmaxf(mx, __shfl_xor(mx, 4, 32));
            mx = fmaxf(mx, __shfl_xor(mx, 8, 32));
            float mold = sm[w][row];
            float mnew = fmaxf(mold, mx);
            float a_   = __expf(mold - mnew);
            float rs = 0.0f;
            #pragma unroll
            for (int nb = 0; nb < 4; ++nb) {
                float pv = __expf(S[nb][r] - mnew);
                rs += pv;
                lp[w][row][nb * 16 + ln] = (f16)pv;
            }
            rs += __shfl_xor(rs, 1, 32);
            rs += __shfl_xor(rs, 2, 32);
            rs += __shfl_xor(rs, 4, 32);
            rs += __shfl_xor(rs, 8, 32);
            if (ln == 0) {
                sm[w][row] = mnew;
                sl[w][row] = sl[w][row] * a_ + rs;
            }
            alpha[r] = a_;
        }

        // rescale O accumulators
        #pragma unroll
        for (int nt = 0; nt < 16; ++nt)
            #pragma unroll
            for (int r = 0; r < 8; ++r)
                O[nt][r] *= alpha[r];

        asm volatile("s_wait_dscnt 0" ::: "memory");   // P scratch visible

        // ---- O += P * V  (V stored transposed -> contiguous B frags) ----
        #pragma unroll
        for (int kc = 0; kc < 64; kc += 32) {
            const f16* pp = &lp[w][ln][kc + half * 16];
            union { v16h v; v8h h[2]; } A;
            A.h[0] = *(const v8h*)pp;
            A.h[1] = *(const v8h*)(pp + 8);
            #pragma unroll
            for (int nt = 0; nt < 16; ++nt) {
                int col = d0 + nt * 16 + ln;
                v16h Bv = *(const v16h*)(VTb + (size_t)col * SS + (kb + kc + half * 16));
                O[nt] = wmma16(A.v, Bv, O[nt]);
            }
        }
    }

    // ---- finalize: ctx = O / l ----
    #pragma unroll
    for (int r = 0; r < 8; ++r) {
        int row = r + 8 * half;
        float inv = 1.0f / sl[w][row];
        size_t gr = (size_t)b * SS + mrow + row;
        #pragma unroll
        for (int nt = 0; nt < 16; ++nt) {
            int col = d0 + nt * 16 + ln;
            ctx[gr * DD + col] = (f16)(O[nt][r] * inv);
        }
    }
}

// ---------------- Kernel 3: out-proj + bias + residual + LayerNorm ----------
// grid (RR/16); block 256 (8 waves). Tile: 16 rows x 512 cols, wave w owns 64 cols.

__global__ __launch_bounds__(256) void outln_kernel(
        const f16* __restrict__ ctx, const f16* __restrict__ WoT,
        const float* __restrict__ bo, const float* __restrict__ x,
        const float* __restrict__ gamma, const float* __restrict__ beta,
        float* __restrict__ out) {
    __shared__ float sh[16][512];   // 32 KB h tile

    int tid = threadIdx.x;
    int w = tid >> 5, l = tid & 31, half = l >> 4, ln = l & 15;
    int mbase = blockIdx.x * 16;
    int nbase = w * 64;

    const v8f zero8 = {0.f,0.f,0.f,0.f,0.f,0.f,0.f,0.f};
    v8f acc[4] = {zero8, zero8, zero8, zero8};

    for (int kb = 0; kb < DD; kb += 32) {
        v16h A = load_a_frag(ctx, DD, mbase + ln, kb, half);
        #pragma unroll
        for (int nb = 0; nb < 4; ++nb) {
            v16h Bv = load_b_frag(WoT, DD, nbase + nb * 16 + ln, kb, half);
            acc[nb] = wmma16(A, Bv, acc[nb]);
        }
    }

    #pragma unroll
    for (int nb = 0; nb < 4; ++nb) {
        int col = nbase + nb * 16 + ln;
        float bsv = bo[col];
        #pragma unroll
        for (int r = 0; r < 8; ++r) {
            int row = r + 8 * half;
            size_t gr = (size_t)(mbase + row);
            sh[row][col] = acc[nb][r] + bsv + x[gr * DD + col];   // + residual
        }
    }
    __syncthreads();

    // LayerNorm: 16 threads per row, 32 elems each
    int row = tid >> 4;
    int c0 = tid & 15;
    const float* hr = sh[row];

    float s = 0.0f;
    #pragma unroll
    for (int i = 0; i < 32; ++i) s += hr[c0 + 16 * i];
    s += __shfl_xor(s, 1, 32);
    s += __shfl_xor(s, 2, 32);
    s += __shfl_xor(s, 4, 32);
    s += __shfl_xor(s, 8, 32);
    float mean = s * (1.0f / 512.0f);

    float v = 0.0f;
    #pragma unroll
    for (int i = 0; i < 32; ++i) { float d = hr[c0 + 16 * i] - mean; v += d * d; }
    v += __shfl_xor(v, 1, 32);
    v += __shfl_xor(v, 2, 32);
    v += __shfl_xor(v, 4, 32);
    v += __shfl_xor(v, 8, 32);
    float rstd = rsqrtf(v * (1.0f / 512.0f) + 1e-3f);

    size_t gr = (size_t)(mbase + row);
    #pragma unroll
    for (int i = 0; i < 32; ++i) {
        int col = c0 + 16 * i;
        out[gr * DD + col] = gamma[col] * (hr[col] - mean) * rstd + beta[col];
    }
}

// ---------------- launch ----------------

extern "C" void kernel_launch(void* const* d_in, const int* in_sizes, int n_in,
                              void* d_out, int out_size, void* d_ws, size_t ws_size,
                              hipStream_t stream) {
    const float* x     = (const float*)d_in[0];
    const float* Wq    = (const float*)d_in[1];
    const float* bq    = (const float*)d_in[2];
    const float* Wk    = (const float*)d_in[3];
    const float* bk    = (const float*)d_in[4];
    const float* Wv    = (const float*)d_in[5];
    const float* bv    = (const float*)d_in[6];
    const float* Wo    = (const float*)d_in[7];
    const float* bo    = (const float*)d_in[8];
    const float* gamma = (const float*)d_in[9];
    const float* beta  = (const float*)d_in[10];
    float* out = (float*)d_out;

    char* ws = (char*)d_ws;
    size_t off = 0;
    const size_t xh_bytes = (size_t)RR * DD * sizeof(f16);   // 16 MiB
    const size_t wt_bytes = (size_t)DD * DD * sizeof(f16);   // 0.5 MiB

    f16* xh   = (f16*)(ws + off); off += xh_bytes;
    f16* WqT  = (f16*)(ws + off); off += wt_bytes;
    f16* WkT  = (f16*)(ws + off); off += wt_bytes;
    f16* WvT  = (f16*)(ws + off); off += wt_bytes;
    f16* WoT  = (f16*)(ws + off); off += wt_bytes;
    f16* Qh   = (f16*)(ws + off); off += xh_bytes;
    f16* Kh   = (f16*)(ws + off); off += xh_bytes;
    f16* VTh  = (f16*)(ws + off); off += xh_bytes;
    f16* ctxh = (f16*)(ws + off); off += xh_bytes;
    (void)ws_size; (void)in_sizes; (void)n_in; (void)out_size;

    // 0a: x -> f16
    cvt_x_kernel<<<dim3((RR * DD) / 256), dim3(256), 0, stream>>>(x, xh, RR * DD);
    // 0b: weights -> transposed f16
    cvt_w_kernel<<<dim3(DD / 16, DD / 16, 4), dim3(16, 16), 0, stream>>>(
        Wq, Wk, Wv, Wo, WqT, WkT, WvT, WoT);
    // 1: Q,K,V projections (Q pre-scaled, V stored transposed)
    qkv_kernel<<<dim3(RR / 64, DD / 128, 3), dim3(256), 0, stream>>>(
        xh, WqT, WkT, WvT, bq, bk, bv, Qh, Kh, VTh);
    // 2: flash attention -> ctx (f16)
    attn_kernel<<<dim3(SS / 64, BB), dim3(256), 0, stream>>>(Qh, Kh, VTh, ctxh);
    // 3: output projection + residual + LayerNorm -> out (f32)
    outln_kernel<<<dim3(RR / 16), dim3(256), 0, stream>>>(
        ctxh, WoT, bo, x, gamma, beta, out);
}
